// DenseHyperbolic_7370163880196
// MI455X (gfx1250) — matrix operand order repaired and checked by reading, not compiled, not run on hardware
//
#include <hip/hip_runtime.h>
#include <math.h>

typedef __attribute__((ext_vector_type(16))) __bf16 v16bf;
typedef __attribute__((ext_vector_type(8)))  float  v8f;

#define EPSV      1e-4f
#define ACOSH_MIN 1.0001f
#define CLIP_MAG  8.0f
#define D 256
#define MROWS 32           // rows per block (2 M-tiles)

__device__ __forceinline__ unsigned short f2bf(float f) {
    unsigned int u = __float_as_uint(f);
    unsigned int r = u + 0x7FFFu + ((u >> 16) & 1u);
    return (unsigned short)(r >> 16);
}

// ---------------------------------------------------------------------------
// Kernel 1: pre-pack fp32 weights [K=256][N=256] into bf16 WMMA B-fragments.
//   lane 0-15 : N = lane,    K = kb*32 + 0..15  (2 bf16/dword, K ascending)
//   lane 16-31: N = lane-16, K = kb*32 + 16..31
// frag dword offset = ((ntile*8 + kb)*32 + lane)*8   (32B contiguous per lane)
// Total 128 KB in d_ws.
// ---------------------------------------------------------------------------
__global__ void prepack_w_bf16(const float* __restrict__ W,
                               unsigned int* __restrict__ frag) {
    int g = blockIdx.x * blockDim.x + threadIdx.x;   // 0..4095
    int lane  = g & 31;
    int kb    = (g >> 5) & 7;
    int ntile = g >> 8;
    int n  = ntile * 16 + (lane & 15);
    int k0 = kb * 32 + ((lane >> 4) << 4);
    unsigned int o[8];
#pragma unroll
    for (int v = 0; v < 8; ++v) {
        float lo = W[(k0 + 2 * v) * D + n];
        float hi = W[(k0 + 2 * v + 1) * D + n];
        o[v] = (unsigned int)f2bf(lo) | ((unsigned int)f2bf(hi) << 16);
    }
    uint4* dst = (uint4*)(frag + (size_t)g * 8);
    dst[0] = make_uint4(o[0], o[1], o[2], o[3]);
    dst[1] = make_uint4(o[4], o[5], o[6], o[7]);
}

// ---------------------------------------------------------------------------
// Kernel 2: bias sum-of-squares, computed once (bias has D-1 = 255 elements).
// ---------------------------------------------------------------------------
__global__ void bias_sq_kernel(const float* __restrict__ bias,
                               float* __restrict__ outv) {
    __shared__ float red[256];
    int t = threadIdx.x;
    float bv = (t < D - 1) ? bias[t] : 0.f;
    red[t] = bv * bv;
    __syncthreads();
    for (int s = 128; s > 0; s >>= 1) {
        if (t < s) red[t] += red[t + s];
        __syncthreads();
    }
    if (t == 0) outv[0] = red[0];
}

// ---------------------------------------------------------------------------
// Kernel 3: fused hyperbolic dense layer.
// 128 threads = 4 waves; 32 rows/block; each wave: 2 M-tiles x 4 N-tiles,
// B-fragment loaded once per (kb,nt) and reused for both M-tiles.
// ---------------------------------------------------------------------------
__global__ __launch_bounds__(128)
void hyper_dense_kernel(const float* __restrict__ vec,
                        const float* __restrict__ cin_p,
                        const float* __restrict__ cout_p,
                        const float* __restrict__ bias_in,
                        const unsigned int* __restrict__ wfrag,
                        const float* __restrict__ bias_sq_p,
                        float* __restrict__ out) {
    __shared__ __align__(16) unsigned short A_lds[MROWS][D];  // bf16 tangent
    __shared__ __align__(16) float C_lds[MROWS][D];           // GEMM output w
    __shared__ float b_lds[D];
    __shared__ float part[MROWS][8][2];
    __shared__ float mults[MROWS];
    __shared__ float rowsc[MROWS][4];

    const int t    = threadIdx.x;
    const int lane = t & 31;
    const int wave = t >> 5;
    const int r    = t >> 3;   // 0..15 (8 threads per row; rows r and r+16)
    const int j    = t & 7;    // column chunk (32 cols)

    const float cin  = cin_p[0];
    const float cout = cout_p[0];
    const float rc   = sqrtf(cin);
    const float rc2  = sqrtf(cout);
    const float bsq  = bias_sq_p[0];

    const long long rowbase = (long long)blockIdx.x * MROWS;

    // -------- bias values -> LDS --------
    {
#pragma unroll
        for (int i = 0; i < 2; ++i) {
            int idx = t * 2 + i;
            b_lds[idx] = (idx == 0) ? 0.f : bias_in[idx - 1];
        }
    }

    // -------- prologue: load 2 rows/thread-group, spatial-norm partials ----
    float va[32], vb[32];
    {
        const float* s0 = vec + (rowbase + r) * D + j * 32;
        const float* s1 = vec + (rowbase + r + 16) * D + j * 32;
        float ssa = 0.f, ssb = 0.f;
#pragma unroll
        for (int i = 0; i < 32; ++i) {
            va[i] = s0[i];
            vb[i] = s1[i];
            if (!(j == 0 && i == 0)) { ssa += va[i] * va[i]; ssb += vb[i] * vb[i]; }
        }
        part[r][j][0] = ssa;
        part[r + 16][j][0] = ssb;
    }
    __syncthreads();

    if (j == 0) {
#pragma unroll
        for (int p = 0; p < 2; ++p) {
            int rr = r + p * 16;
            float s2 = 0.f;
#pragma unroll
            for (int i = 0; i < 8; ++i) s2 += part[rr][i][0];
            float s    = sqrtf(s2);
            float v0p  = sqrtf(cin + s2);                    // projected time
            float dist = rc * acoshf(fmaxf(v0p / rc - EPSV, ACOSH_MIN));
            mults[rr]  = dist / (s + EPSV);                  // logmap multiplier
        }
    }
    __syncthreads();

    {
        float m0 = mults[r], m1 = mults[r + 16];
#pragma unroll
        for (int i = 0; i < 32; ++i) {
            int col = j * 32 + i;
            A_lds[r][col]      = f2bf((col == 0) ? 0.f : m0 * va[i]);
            A_lds[r + 16][col] = f2bf((col == 0) ? 0.f : m1 * vb[i]);
        }
    }
    __syncthreads();

    // -------- GEMM: tangent[32x256] @ W[256x256] via bf16 WMMA --------
    v8f acc[2][4];
#pragma unroll
    for (int mt = 0; mt < 2; ++mt)
#pragma unroll
        for (int nt = 0; nt < 4; ++nt)
            acc[mt][nt] = (v8f){0,0,0,0,0,0,0,0};

    const int mA    = lane & 15;
    const int halfA = lane >> 4;
#pragma unroll
    for (int kb = 0; kb < 8; ++kb) {
        // A fragments for both M-tiles (16-bit A 16x32 layout)
        union { uint4 u[2]; v16bf bv; } a0, a1;
        const unsigned short* ap0 = &A_lds[mA][kb * 32 + halfA * 16];
        const unsigned short* ap1 = &A_lds[mA + 16][kb * 32 + halfA * 16];
        a0.u[0] = *(const uint4*)(ap0);
        a0.u[1] = *(const uint4*)(ap0 + 8);
        a1.u[0] = *(const uint4*)(ap1);
        a1.u[1] = *(const uint4*)(ap1 + 8);
#pragma unroll
        for (int nt = 0; nt < 4; ++nt) {
            int ntile = wave * 4 + nt;
            union { uint4 u[2]; v16bf bv; } bfr;
            const uint4* bp =
                (const uint4*)(wfrag + ((size_t)(ntile * 8 + kb) * 32 + lane) * 8);
            bfr.u[0] = bp[0];
            bfr.u[1] = bp[1];
            acc[0][nt] = __builtin_amdgcn_wmma_f32_16x16x32_bf16(
                false, a0.bv, false, bfr.bv, (short)0, acc[0][nt], false, false);
            acc[1][nt] = __builtin_amdgcn_wmma_f32_16x16x32_bf16(
                false, a1.bv, false, bfr.bv, (short)0, acc[1][nt], false, false);
        }
    }

    // -------- scatter C to LDS (mask column 0) --------
    {
        int col16  = lane & 15;
        int rowoff = (lane >> 4) << 3;
#pragma unroll
        for (int mt = 0; mt < 2; ++mt)
#pragma unroll
            for (int nt = 0; nt < 4; ++nt) {
                int col = (wave * 4 + nt) * 16 + col16;
#pragma unroll
                for (int g = 0; g < 8; ++g)
                    C_lds[mt * 16 + rowoff + g][col] =
                        (col == 0) ? 0.f : acc[mt][nt][g];
            }
    }
    __syncthreads();

    // -------- epilogue reductions: R1 = sum w^2, R2 = sum w*b --------
    {
#pragma unroll
        for (int p = 0; p < 2; ++p) {
            int rr = r + p * 16;
            float s1 = 0.f, s2 = 0.f;
#pragma unroll
            for (int i = 0; i < 32; ++i) {
                int col = j * 32 + i;
                float wv = C_lds[rr][col];
                s1 += wv * wv;
                s2 += wv * b_lds[col];
            }
            part[rr][j][0] = s1;
            part[rr][j][1] = s2;
        }
    }
    __syncthreads();

    if (j == 0) {
#pragma unroll
        for (int p = 0; p < 2; ++p) {
            int rr = r + p * 16;
            float R1 = 0.f, R2 = 0.f;
#pragma unroll
            for (int i = 0; i < 8; ++i) { R1 += part[rr][i][0]; R2 += part[rr][i][1]; }

            // expmap(w, origin): h_i = shn*w_i, h0 = rc*cosh(n1)
            float n1  = sqrtf(R1) / rc + EPSV;
            float n1c = fminf(n1, CLIP_MAG);
            float ch1 = coshf(n1c);
            float shn = sinhf(n1c) / n1;
            float sh2 = shn * shn * R1;            // spatial |h|^2
            float h0p = sqrtf(cin + sh2);          // re-projected time coord
            // distance(origin, h) and bias-transport scalar q
            float distO = rc * acoshf(fmaxf(ch1, ACOSH_MIN));
            float q = R2 / (distO * distO);
            // logmap(origin, h)
            float dist2 = rc * acoshf(fmaxf(h0p / rc - EPSV, ACOSH_MIN));
            float u0    = rc - h0p * h0p / rc;
            float un    = sqrtf(fmaxf((h0p * h0p / cin) * sh2 - u0 * u0, 0.f));
            float mult2 = dist2 / (un + EPSV);
            float S0    = mult2 * u0;
            float beta  = 1.f - mult2 * (h0p / rc) * shn;   // S_i = beta*w_i
            float t0    = -q * S0;
            float qb    = q * beta;                         // t_i = b_i - qb*w_i
            float T2    = t0 * t0 + bsq - 2.f * qb * R2 + qb * qb * R1;
            // expmap(t, h): biased_i = epv*w_i + dlt*b_i
            float n3  = sqrtf(T2) / rc + EPSV;
            float n3c = fminf(n3, CLIP_MAG);
            float ch3 = coshf(n3c);
            float dlt = sinhf(n3c) / n3;
            float epv = ch3 * shn - dlt * qb;
            float sb  = epv * epv * R1 + 2.f * epv * dlt * R2 + dlt * dlt * bsq;
            // logmap(biased, origin): bt_i = mult4*biased_i
            float b0p   = sqrtf(cin + sb);
            float dist4 = rc * acoshf(fmaxf(b0p / rc - EPSV, ACOSH_MIN));
            float mult4 = dist4 / (sqrtf(sb) + EPSV);
            // final expmap at out_origin2
            float n5  = sqrtf(mult4 * mult4 * sb) / rc2 + EPSV;
            float n5c = fminf(n5, CLIP_MAG);
            float kap = (sinhf(n5c) / n5) * mult4;
            rowsc[rr][0] = rc2 * coshf(n5c);
            rowsc[rr][1] = kap * epv;
            rowsc[rr][2] = kap * dlt;
        }
    }
    __syncthreads();

    // -------- elementwise final write --------
    {
#pragma unroll
        for (int p = 0; p < 2; ++p) {
            int rr = r + p * 16;
            float o0 = rowsc[rr][0], ke = rowsc[rr][1], kd = rowsc[rr][2];
            float* dst = out + (rowbase + rr) * D + j * 32;
#pragma unroll
            for (int i = 0; i < 32; ++i) {
                int col = j * 32 + i;
                dst[i] = (col == 0) ? o0 : ke * C_lds[rr][col] + kd * b_lds[col];
            }
        }
    }
}

extern "C" void kernel_launch(void* const* d_in, const int* in_sizes, int n_in,
                              void* d_out, int out_size, void* d_ws, size_t ws_size,
                              hipStream_t stream) {
    const float* vec  = (const float*)d_in[0];
    const float* cin  = (const float*)d_in[1];
    const float* cout = (const float*)d_in[2];
    const float* W    = (const float*)d_in[3];
    const float* bias = (const float*)d_in[4];
    float* outp       = (float*)d_out;

    unsigned int* wfrag = (unsigned int*)d_ws;                 // 128 KB
    float* bias_sq      = (float*)((char*)d_ws + 131072);      // +4 B

    int B = in_sizes[0] / D;          // 131072 rows
    prepack_w_bf16<<<16, 256, 0, stream>>>(W, wfrag);
    bias_sq_kernel<<<1, 256, 0, stream>>>(bias, bias_sq);
    hyper_dense_kernel<<<B / MROWS, 128, 0, stream>>>(vec, cin, cout, bias,
                                                      wfrag, bias_sq, outp);
}
